// EWGCN_50474455662617
// MI455X (gfx1250) — matrix operand order repaired and checked by reading, compiled, not run on hardware
//
#include <hip/hip_runtime.h>

#define H     128
#define EMBED 300
#define NCLS  20

typedef __attribute__((ext_vector_type(16))) _Float16 v16h;
typedef __attribute__((ext_vector_type(8)))  _Float16 v8h;
typedef __attribute__((ext_vector_type(4)))  _Float16 v4h;
typedef __attribute__((ext_vector_type(8)))  float    v8f;

// Guaranteed-native FP32 atomic add (no CAS-loop fallback). No-return form:
// tracked with STOREcnt; kernel-end implicit S_WAIT_IDLE covers completion.
__device__ __forceinline__ void atomAddF(float* p, float v) {
  asm volatile("global_atomic_add_f32 %0, %1, off"
               :
               : "v"(p), "v"(v)
               : "memory");
}

// ---------------------------------------------------------------------------
// Weight prep: W1 [300,128]f32 -> W1T [128,320]f16 (N-major, K zero-padded)
//              W2 [128,128]f32 -> W2T [128,128]f16 (N-major)
// ---------------------------------------------------------------------------
__global__ void gcn_prep_w(const float* __restrict__ W1, const float* __restrict__ W2,
                           _Float16* __restrict__ W1T, _Float16* __restrict__ W2T) {
  int t = blockIdx.x * blockDim.x + threadIdx.x;
  if (t < H * 320) {
    int n = t / 320, k = t % 320;
    W1T[t] = (_Float16)((k < EMBED) ? W1[(size_t)k * H + n] : 0.0f);
  } else {
    int u = t - H * 320;
    if (u < H * H) {
      int n = u / H, k = u % H;
      W2T[u] = (_Float16)W2[(size_t)k * H + n];
    }
  }
}

// ---------------------------------------------------------------------------
// Degree / dinv
// ---------------------------------------------------------------------------
__global__ void gcn_fill(float* __restrict__ p, float val, int n) {
  int i = blockIdx.x * blockDim.x + threadIdx.x;
  if (i < n) p[i] = val;
}

__global__ void gcn_edge_deg(const int* __restrict__ dst, float* __restrict__ deg, int E) {
  int i = blockIdx.x * blockDim.x + threadIdx.x;
  if (i < E) atomAddF(&deg[dst[i]], 1.0f);
}

__global__ void gcn_rsqrt(float* __restrict__ p, int n) {
  int i = blockIdx.x * blockDim.x + threadIdx.x;
  if (i < n) p[i] = rsqrtf(p[i]);
}

// ---------------------------------------------------------------------------
// WMMA GEMM: C[M,128] = A[M,LDA](f16) x B (BT is N-major f16 [128, KTILES*32])
// One workgroup (8 waves) = one 16-row M tile x full N=128.
// ---------------------------------------------------------------------------
template <int KTILES, int LDA, bool GUARD>
__global__ __launch_bounds__(256)
void gcn_gemm_wmma(const _Float16* __restrict__ A, const _Float16* __restrict__ BT,
                   float* __restrict__ C, int M) {
  const int lane = threadIdx.x & 31;
  const int wv   = threadIdx.x >> 5;      // N tile 0..7
  const int hi   = lane >> 4;             // half-wave select
  const int l15  = lane & 15;

  int row = blockIdx.x * 16 + l15;
  if (row >= M) row = M - 1;              // clamp reads (stores guarded below)
  const _Float16* __restrict__ arow = A + (size_t)row * LDA;

  const int LDB = KTILES * 32;
  const int col = wv * 16 + l15;
  const _Float16* __restrict__ bcol = BT + (size_t)col * LDB;

  v8f c = {};
#pragma unroll
  for (int kt = 0; kt < KTILES; ++kt) {
    // ---- A fragment: 16x32 f16. lane-group hi: K groups {ka..ka+7, ka+16..ka+23}
    const int ka = kt * 32 + hi * 8;
    v16h a;
    if (GUARD && kt == KTILES - 1) {
#pragma unroll
      for (int e = 0; e < 8; ++e) {
        int k1 = ka + e, k2 = ka + 16 + e;
        a[e]     = (k1 < LDA) ? arow[k1] : (_Float16)0.0f;
        a[e + 8] = (k2 < LDA) ? arow[k2] : (_Float16)0.0f;
      }
    } else {
      const v4h* p = reinterpret_cast<const v4h*>(arow + ka);        // 8B aligned
      const v4h* q = reinterpret_cast<const v4h*>(arow + ka + 16);
      v4h x0 = p[0], x1 = p[1], x2 = q[0], x3 = q[1];
#pragma unroll
      for (int j = 0; j < 4; ++j) {
        a[j] = x0[j]; a[4 + j] = x1[j]; a[8 + j] = x2[j]; a[12 + j] = x3[j];
      }
    }
    // ---- B fragment: 32x16 f16, col = lane&15, K groups {kb..kb+7, kb+16..kb+23}
    const int kb = kt * 32 + hi * 8;
    const v8h* bq = reinterpret_cast<const v8h*>(bcol + kb);         // 16B aligned
    v8h y0 = bq[0], y1 = bq[2];
    v16h b;
#pragma unroll
    for (int j = 0; j < 8; ++j) { b[j] = y0[j]; b[8 + j] = y1[j]; }

    c = __builtin_amdgcn_wmma_f32_16x16x32_f16(false, a, false, b, (short)0, c,
                                               false, false);
  }

  // D layout: VGPR v -> row (v + 8*hi), col = lane&15
  const int m0 = blockIdx.x * 16 + hi * 8;
#pragma unroll
  for (int v = 0; v < 8; ++v) {
    int m = m0 + v;
    if (m < M) C[(size_t)m * H + col] = c[v];
  }
}

// ---------------------------------------------------------------------------
// Aggregation: agg = self-loop term, then per-edge scatter-add
// ---------------------------------------------------------------------------
__global__ void gcn_self_loop(const float* __restrict__ h, const float* __restrict__ dinv,
                              float* __restrict__ agg, int total) {
  int t = blockIdx.x * blockDim.x + threadIdx.x;
  if (t < total) {
    int r = t >> 7;
    float dv = dinv[r];
    agg[t] = h[t] * dv * dv;
  }
}

__global__ __launch_bounds__(256)
void gcn_edge_agg(const int* __restrict__ src, const int* __restrict__ dst,
                  const float* __restrict__ dinv, const float* __restrict__ h,
                  float* __restrict__ agg, int E) {
  int w    = (blockIdx.x * blockDim.x + threadIdx.x) >> 5;  // wave per edge
  int lane = threadIdx.x & 31;
  if (w >= E) return;
  int s = src[w], d = dst[w];
  float nrm = dinv[s] * dinv[d];
  const float4 hv = *reinterpret_cast<const float4*>(h + (size_t)s * H + lane * 4);
  float* o = agg + (size_t)d * H + lane * 4;
  atomAddF(o + 0, hv.x * nrm);
  atomAddF(o + 1, hv.y * nrm);
  atomAddF(o + 2, hv.z * nrm);
  atomAddF(o + 3, hv.w * nrm);
}

__global__ void gcn_bias_relu_f16(const float* __restrict__ agg, const float* __restrict__ b,
                                  _Float16* __restrict__ o, int total) {
  int t = blockIdx.x * blockDim.x + threadIdx.x;
  if (t < total) o[t] = (_Float16)fmaxf(agg[t] + b[t & (H - 1)], 0.0f);
}

__global__ void gcn_bias_relu_f32(float* __restrict__ agg, const float* __restrict__ b, int total) {
  int t = blockIdx.x * blockDim.x + threadIdx.x;
  if (t < total) agg[t] = fmaxf(agg[t] + b[t & (H - 1)], 0.0f);
}

// ---------------------------------------------------------------------------
// Pooling
// ---------------------------------------------------------------------------
__global__ void gcn_zero_pool(float* __restrict__ ent_sum, float* __restrict__ node_sum,
                              float* __restrict__ ent_cnt, float* __restrict__ node_cnt,
                              int GH, int G) {
  int t = blockIdx.x * blockDim.x + threadIdx.x;
  if (t < GH) { ent_sum[t] = 0.0f; node_sum[t] = 0.0f; }
  if (t < G)  { ent_cnt[t] = 0.0f; node_cnt[t] = 0.0f; }
}

__global__ __launch_bounds__(256)
void gcn_pool(const float* __restrict__ x, const int* __restrict__ batch,
              const unsigned char* __restrict__ mask,
              float* __restrict__ ent_sum, float* __restrict__ node_sum,
              float* __restrict__ ent_cnt, float* __restrict__ node_cnt, int N) {
  int w    = (blockIdx.x * blockDim.x + threadIdx.x) >> 5;  // wave per node
  int lane = threadIdx.x & 31;
  if (w >= N) return;
  int  b = batch[w];
  bool m = mask[w] != 0;
  const float4 xv = *reinterpret_cast<const float4*>(x + (size_t)w * H + lane * 4);
  float* ns = node_sum + (size_t)b * H + lane * 4;
  atomAddF(ns + 0, xv.x); atomAddF(ns + 1, xv.y);
  atomAddF(ns + 2, xv.z); atomAddF(ns + 3, xv.w);
  if (m) {
    float* es = ent_sum + (size_t)b * H + lane * 4;
    atomAddF(es + 0, xv.x); atomAddF(es + 1, xv.y);
    atomAddF(es + 2, xv.z); atomAddF(es + 3, xv.w);
  }
  if (lane == 0) {
    atomAddF(&node_cnt[b], 1.0f);
    if (m) atomAddF(&ent_cnt[b], 1.0f);
  }
}

// ---------------------------------------------------------------------------
// Head: doc_vec select + [1,128]x[128,20] + log_softmax, one wave per graph
// ---------------------------------------------------------------------------
__global__ __launch_bounds__(32)
void gcn_head(const float* __restrict__ ent_sum, const float* __restrict__ ent_cnt,
              const float* __restrict__ node_sum, const float* __restrict__ node_cnt,
              const float* __restrict__ Wout, const float* __restrict__ bout,
              float* __restrict__ out) {
  __shared__ float dv[H];
  __shared__ float lg[NCLS];
  const int g = blockIdx.x, lane = threadIdx.x;
  const float ec = ent_cnt[g];
  const bool  fb = ec < 1e-6f;
  const float nc = fmaxf(node_cnt[g], 1.0f);
  const float inv = fb ? (1.0f / nc) : (1.0f / (ec + 1e-6f));
  const float* sum = fb ? (node_sum + (size_t)g * H) : (ent_sum + (size_t)g * H);
  for (int f = lane; f < H; f += 32) dv[f] = sum[f] * inv;
  __syncthreads();
  if (lane < NCLS) {
    float acc = bout[lane];
    for (int k = 0; k < H; ++k) acc += dv[k] * Wout[k * NCLS + lane];
    lg[lane] = acc;
  }
  __syncthreads();
  if (lane < NCLS) {
    float mx = lg[0];
#pragma unroll
    for (int c = 1; c < NCLS; ++c) mx = fmaxf(mx, lg[c]);
    float s = 0.0f;
#pragma unroll
    for (int c = 0; c < NCLS; ++c) s += __expf(lg[c] - mx);
    out[g * NCLS + lane] = lg[lane] - mx - __logf(s);
  }
}

// ---------------------------------------------------------------------------
extern "C" void kernel_launch(void* const* d_in, const int* in_sizes, int n_in,
                              void* d_out, int out_size, void* d_ws, size_t ws_size,
                              hipStream_t stream) {
  const _Float16*      x     = (const _Float16*)d_in[0];       // [N,300] f16
  const int*           ei    = (const int*)d_in[1];            // [2,E]
  const int*           batch = (const int*)d_in[2];            // [N]
  const unsigned char* mask  = (const unsigned char*)d_in[3];  // [N] bool
  const float* W1   = (const float*)d_in[4];
  const float* b1   = (const float*)d_in[5];
  const float* W2   = (const float*)d_in[6];
  const float* b2   = (const float*)d_in[7];
  const float* Wout = (const float*)d_in[8];
  const float* bout = (const float*)d_in[9];
  float*       out  = (float*)d_out;

  const int N = in_sizes[2];        // 100000
  const int E = in_sizes[1] / 2;    // 1.6M
  const int G = out_size / NCLS;    // 512

  // ---- workspace carve-up (256B aligned) ----
  char*  ws  = (char*)d_ws;
  size_t off = 0;
  auto take = [&](size_t bytes) -> void* {
    void* p = ws + off;
    off = (off + bytes + 255) & ~(size_t)255;
    return p;
  };
  float*    hbuf     = (float*)take((size_t)N * H * 4);
  float*    agg      = (float*)take((size_t)N * H * 4);
  _Float16* h1       = (_Float16*)take((size_t)N * H * 2);
  float*    dinv     = (float*)take((size_t)N * 4);       // deg, then rsqrt in place
  float*    ent_sum  = (float*)take((size_t)G * H * 4);
  float*    node_sum = (float*)take((size_t)G * H * 4);
  float*    ent_cnt  = (float*)take((size_t)G * 4);
  float*    node_cnt = (float*)take((size_t)G * 4);
  _Float16* W1T      = (_Float16*)take((size_t)H * 320 * 2);
  _Float16* W2T      = (_Float16*)take((size_t)H * H * 2);

  const int* src = ei;
  const int* dst = ei + E;

  const int NH      = N * H;
  const int mtiles  = (N + 15) / 16;
  const int blkN    = (N + 255) / 256;
  const int blkE    = (E + 255) / 256;
  const int blkNH   = (NH + 255) / 256;
  const int blkEw   = (E * 4 + 31) / 32;   // wave(32) per edge, 256 thr/blk => E/8
  const int blkNw   = (N * 4 + 31) / 32;   // wave per node
  const int blkPrep = (H * 320 + H * H + 255) / 256;

  // degrees -> dinv
  gcn_prep_w<<<blkPrep, 256, 0, stream>>>(W1, W2, W1T, W2T);
  gcn_fill<<<blkN, 256, 0, stream>>>(dinv, 1.0f, N);        // self-loop
  gcn_edge_deg<<<blkE, 256, 0, stream>>>(dst, dinv, E);
  gcn_rsqrt<<<blkN, 256, 0, stream>>>(dinv, N);

  // ---- layer 1 ----
  gcn_gemm_wmma<10, EMBED, true><<<mtiles, 256, 0, stream>>>(x, W1T, hbuf, N);
  gcn_self_loop<<<blkNH, 256, 0, stream>>>(hbuf, dinv, agg, NH);
  gcn_edge_agg<<<blkEw, 256, 0, stream>>>(src, dst, dinv, hbuf, agg, E);
  gcn_bias_relu_f16<<<blkNH, 256, 0, stream>>>(agg, b1, h1, NH);

  // ---- layer 2 ----
  gcn_gemm_wmma<4, H, false><<<mtiles, 256, 0, stream>>>(h1, W2T, hbuf, N);
  gcn_self_loop<<<blkNH, 256, 0, stream>>>(hbuf, dinv, agg, NH);
  gcn_edge_agg<<<blkEw, 256, 0, stream>>>(src, dst, dinv, hbuf, agg, E);
  gcn_bias_relu_f32<<<blkNH, 256, 0, stream>>>(agg, b2, NH);

  // ---- pooling + head ----
  gcn_zero_pool<<<(G * H + 255) / 256, 256, 0, stream>>>(ent_sum, node_sum,
                                                         ent_cnt, node_cnt, G * H, G);
  gcn_pool<<<blkNw, 256, 0, stream>>>(agg, batch, mask, ent_sum, node_sum,
                                      ent_cnt, node_cnt, N);
  gcn_head<<<G, 32, 0, stream>>>(ent_sum, ent_cnt, node_sum, node_cnt, Wout, bout, out);
}
